// NVFP4FakeQuantizedMoE_39779987095894
// MI455X (gfx1250) — matrix operand order repaired and checked by reading, not compiled, not run on hardware
//
#include <hip/hip_runtime.h>
#include <hip/hip_bf16.h>
#include <math.h>

namespace {

constexpr int kE = 8, kT = 1024, kTopK = 2, kH = 1024, kI = 1024;
constexpr int kP = kT * kTopK;        // 2048 expanded rows
constexpr int kCap = 4096;            // padded row capacity (>= 2048 + 8*63)
constexpr int kNT1 = (2 * kI) / 16;   // 128 n-tiles per k-tile in gemm1 B
constexpr int kNT2 = kH / 16;         // 64 n-tiles per k-tile in gemm2 B

typedef __attribute__((ext_vector_type(16))) __bf16 v16bf;
typedef __attribute__((ext_vector_type(8)))  __bf16 v8bf;
typedef __attribute__((ext_vector_type(8)))  float  v8f;

// ---------------- workspace layout (bytes) ----------------
constexpr size_t OFF_AMAX = 0;                                  // float[32]: 0=x, 1..8=w1, 9..16=w2, 17=act
constexpr size_t OFF_INTS = 128;                                // int[32]: [0..7]=cnt, [8..15]=segoff, [16]=total
constexpr size_t OFF_ROWT = 256;                                // int[kCap]   expanded-row index per slot
constexpr size_t OFF_ROWW = OFF_ROWT + (size_t)kCap * 4;        // float[kCap] routing weight per slot
constexpr size_t OFF_XS   = OFF_ROWW + (size_t)kCap * 4;        // bf16[kT*kH]      quantized x (row major, A layout)
constexpr size_t OFF_B1   = OFF_XS + (size_t)kT * kH * 2;       // bf16[kE*2I*H]    w1, B-frag swizzled
constexpr size_t OFF_B2   = OFF_B1 + (size_t)kE * 2 * kI * kH * 2; // bf16[kE*H*I]  w2, B-frag swizzled
constexpr size_t OFF_HB   = OFF_B2 + (size_t)kE * kH * kI * 2;  // float[kCap*kI]   silu(gate)*value
constexpr size_t OFF_A2   = OFF_HB + (size_t)kCap * kI * 4;     // bf16[kCap*kI]    quantized act
constexpr size_t WS_NEED  = OFF_A2 + (size_t)kCap * kI * 2;     // ~77.6 MB

// ---------------- helpers ----------------
__device__ inline float bfr(float x) { return (float)(__bf16)x; }

__device__ inline float qdq_e4m3(float v) {          // e4m3fn quant-dequant, v >= 0
  v = fabsf(v);
  if (!(v > 0.f)) return 0.f;                        // 0 / NaN -> 0
  v = fminf(v, 448.f);
  int e; (void)frexpf(v, &e);                        // v in [2^(e-1), 2^e)
  int E = e - 1; if (E < -6) E = -6;                 // denormal spacing floor 2^-9
  float step = ldexpf(1.f, E - 3);
  return fminf(rintf(v / step) * step, 448.f);
}

__device__ inline float round_e2m1(float m) {
  return (m <= 0.25f) ? 0.f :
         (m <  0.75f) ? 0.5f :
         (m <= 1.25f) ? 1.f :
         (m <  1.75f) ? 1.5f :
         (m <= 2.5f ) ? 2.f :
         (m <  3.5f ) ? 3.f :
         (m <= 5.0f ) ? 4.f : 6.f;
}

// quantize one 16-element block; emit s = sign*m*sf  (bf16-exact; dq = s/gsf)
__device__ inline void nvfp4_block(const float* __restrict__ src, float gsf, __bf16* o) {
  float v[16]; float bmax = 0.f;
  const float4* p4 = (const float4*)src;
#pragma unroll
  for (int q = 0; q < 4; ++q) {
    float4 f = p4[q];
    v[q*4+0]=f.x; v[q*4+1]=f.y; v[q*4+2]=f.z; v[q*4+3]=f.w;
  }
#pragma unroll
  for (int i = 0; i < 16; ++i) {
    v[i] = bfr(v[i]);
    float a = fabsf(v[i]);
    if (a == a) bmax = fmaxf(bmax, a);
  }
  float sf = qdq_e4m3(bmax * gsf * (1.f / 6.f));
  if (sf > 0.f) {
    float scale = gsf / sf;
#pragma unroll
    for (int i = 0; i < 16; ++i)
      o[i] = (__bf16)(copysignf(round_e2m1(fabsf(v[i]) * scale), v[i]) * sf);
  } else {
#pragma unroll
    for (int i = 0; i < 16; ++i) o[i] = (__bf16)0.f;
  }
}

__device__ inline void store16(__bf16* dst, const __bf16* s) {
  ((v8bf*)dst)[0] = *(const v8bf*)s;
  ((v8bf*)dst)[1] = *(const v8bf*)(s + 8);
}

// A fragment (16x32 bf16): lane L holds row m=L&15, K = koff..koff+15, koff=(L>>4)*16.
// p must already point at row + koff (+ kt*32 added by caller).
__device__ inline v16bf load_a(const __bf16* p) {
  union { v16bf v; v8bf h[2]; } u;
  u.h[0] = *(const v8bf*)p;
  u.h[1] = *(const v8bf*)(p + 8);
  return u.v;
}

// B fragment from swizzled layout: 512 bf16 per (kt,nt) fragment, lane L reads 16 contiguous.
__device__ inline v16bf load_b(const __bf16* Bbase, int kt, int nt, int NT, int lane) {
  const __bf16* p = Bbase + ((size_t)(kt * NT + nt) * 32 + lane) * 16;
  union { v16bf v; v8bf h[2]; } u;
  u.h[0] = *(const v8bf*)p;
  u.h[1] = *(const v8bf*)(p + 8);
  return u.v;
}

__device__ inline v8f wmma_bf16(v16bf a, v16bf b, v8f c) {
  return __builtin_amdgcn_wmma_f32_16x16x32_bf16(false, a, false, b, (short)0, c, false, false);
}

__device__ inline v8f zero8() { v8f z; for (int i = 0; i < 8; ++i) z[i] = 0.f; return z; }

// ---------------- kernels ----------------
__global__ __launch_bounds__(256) void k_init(float* out, float* amax) {
  int i = blockIdx.x * 256 + threadIdx.x;
  if (i < kT * kH) out[i] = 0.f;
  if (blockIdx.x == 0 && threadIdx.x < 32) amax[threadIdx.x] = 0.f;
}

__global__ __launch_bounds__(256) void k_amax(const float* __restrict__ src, int perSeg,
                                              int blocksPerSeg, float* amax_dst) {
  int seg = blockIdx.x / blocksPerSeg, bin = blockIdx.x % blocksPerSeg;
  int chunk = perSeg / blocksPerSeg;
  const float* p = src + (size_t)seg * perSeg + (size_t)bin * chunk;
  float lm = 0.f;
  for (int i = threadIdx.x; i < chunk; i += 256) {
    float v = fabsf(bfr(p[i]));
    if (v == v) lm = fmaxf(lm, v);
  }
  __shared__ float red[256];
  red[threadIdx.x] = lm; __syncthreads();
  for (int s = 128; s > 0; s >>= 1) {
    if (threadIdx.x < s) red[threadIdx.x] = fmaxf(red[threadIdx.x], red[threadIdx.x + s]);
    __syncthreads();
  }
  if (threadIdx.x == 0)
    atomicMax((unsigned int*)(amax_dst + seg), __float_as_uint(red[0]));
}

// deterministic per-expert compaction: thread e scans rows in order
__global__ void k_route(const int* __restrict__ ri, const float* __restrict__ rw,
                        int* ints, int* rowt, float* roww) {
  __shared__ int cnt[kE], pad[kE], off[kE + 1];
  int tid = threadIdx.x;
  if (tid < kE) {
    int c = 0;
    for (int p = 0; p < kP; ++p) if (ri[p] == tid) c++;
    cnt[tid] = c;
  }
  __syncthreads();
  if (tid == 0) {
    int o = 0;
    for (int e = 0; e < kE; ++e) { off[e] = o; pad[e] = (cnt[e] + 63) & ~63; o += pad[e]; }
    off[kE] = o;
    for (int e = 0; e < kE; ++e) { ints[e] = cnt[e]; ints[8 + e] = off[e]; }
    ints[16] = off[kE];
  }
  __syncthreads();
  if (tid < kE) {
    int j = off[tid];
    for (int p = 0; p < kP; ++p)
      if (ri[p] == tid) {
        int t = p / kTopK;
        rowt[j] = p;
        roww[j] = rw[t * kE + tid];
        ++j;
      }
    for (; j < off[tid] + pad[tid]; ++j) { rowt[j] = 0; roww[j] = 0.f; }
  }
}

__global__ __launch_bounds__(256) void k_qx(const float* __restrict__ x,
                                            const float* __restrict__ amax, __bf16* xs) {
  int id = blockIdx.x * 256 + threadIdx.x;
  if (id >= kT * kH / 16) return;
  float gsf = 2688.f / amax[0];
  __bf16 o[16] __attribute__((aligned(32)));
  nvfp4_block(x + (size_t)id * 16, gsf, o);
  store16(xs + (size_t)id * 16, o);
}

__global__ __launch_bounds__(256) void k_qw1(const float* __restrict__ w,
                                             const float* __restrict__ amax, __bf16* B1) {
  int id = blockIdx.x * 256 + threadIdx.x;            // (e, hb, o), o fastest
  if (id >= kE * (kH / 16) * (2 * kI)) return;
  int o  = id % (2 * kI);
  int r  = id / (2 * kI);
  int hb = r % (kH / 16);
  int e  = r / (kH / 16);
  float gsf = 2688.f / amax[1 + e];
  __bf16 s[16] __attribute__((aligned(32)));
  nvfp4_block(w + ((size_t)(e * 2 * kI + o)) * kH + hb * 16, gsf, s);
  int kt = hb >> 1, lane = ((hb & 1) << 4) | (o & 15), nt = o >> 4;
  size_t off = (size_t)e * 2 * kI * kH + ((size_t)(kt * kNT1 + nt) * 32 + lane) * 16;
  store16(B1 + off, s);
}

__global__ __launch_bounds__(256) void k_qw2(const float* __restrict__ w,
                                             const float* __restrict__ amax, __bf16* B2) {
  int id = blockIdx.x * 256 + threadIdx.x;            // (e, ib, h), h fastest
  if (id >= kE * (kI / 16) * kH) return;
  int h  = id % kH;
  int r  = id / kH;
  int ib = r % (kI / 16);
  int e  = r / (kI / 16);
  float gsf = 2688.f / amax[9 + e];
  __bf16 s[16] __attribute__((aligned(32)));
  nvfp4_block(w + ((size_t)(e * kH + h)) * kI + ib * 16, gsf, s);
  int kt = ib >> 1, lane = ((ib & 1) << 4) | (h & 15), nt = h >> 4;
  size_t off = (size_t)e * kH * kI + ((size_t)(kt * kNT2 + nt) * 32 + lane) * 16;
  store16(B2 + off, s);
}

// gemm1: per wave 32 rows x (32 value cols + 32 gate cols), fused silu*value + act amax
__global__ __launch_bounds__(256) void k_gemm1(const __bf16* __restrict__ Xs,
                                               const __bf16* __restrict__ B1,
                                               const int* __restrict__ ints,
                                               const int* __restrict__ rowt,
                                               float* amax, float* __restrict__ Hbuf) {
  __shared__ int s_off[9];
  int tid = threadIdx.x;
  if (tid < 9) s_off[tid] = (tid < 8) ? ints[8 + tid] : ints[16];
  __syncthreads();
  int mbase = blockIdx.y * 64;
  if (mbase >= s_off[8]) return;
  int e = 0;
  while (e < 7 && mbase >= s_off[e + 1]) ++e;
  int cnt_e = ints[e], segstart = s_off[e];

  int lane = tid & 31, w = tid >> 5;
  int wm = w & 1, wn = w >> 1;
  int m0 = mbase + wm * 32;
  int n0 = blockIdx.x * 128 + wn * 32;

  int rA   = m0 + (lane & 15);
  int tok0 = rowt[rA]      >> 1;       // token = p / TOPK
  int tok1 = rowt[rA + 16] >> 1;
  int koff = (lane >> 4) << 4;
  const __bf16* a0p = Xs + (size_t)tok0 * kH + koff;
  const __bf16* a1p = Xs + (size_t)tok1 * kH + koff;
  const __bf16* Bb  = B1 + (size_t)e * 2 * kI * kH;
  int ntv = n0 >> 4;

  v8f av[2][2], ag[2][2];
#pragma unroll
  for (int i = 0; i < 2; ++i)
#pragma unroll
    for (int j = 0; j < 2; ++j) { av[i][j] = zero8(); ag[i][j] = zero8(); }

  for (int kt = 0; kt < kH / 32; ++kt) {
    v16bf a0  = load_a(a0p + kt * 32);
    v16bf a1  = load_a(a1p + kt * 32);
    v16bf bv0 = load_b(Bb, kt, ntv,      kNT1, lane);
    v16bf bv1 = load_b(Bb, kt, ntv + 1,  kNT1, lane);
    v16bf bg0 = load_b(Bb, kt, 64 + ntv,     kNT1, lane);
    v16bf bg1 = load_b(Bb, kt, 64 + ntv + 1, kNT1, lane);
    av[0][0] = wmma_bf16(a0, bv0, av[0][0]);
    av[0][1] = wmma_bf16(a0, bv1, av[0][1]);
    av[1][0] = wmma_bf16(a1, bv0, av[1][0]);
    av[1][1] = wmma_bf16(a1, bv1, av[1][1]);
    ag[0][0] = wmma_bf16(a0, bg0, ag[0][0]);
    ag[0][1] = wmma_bf16(a0, bg1, ag[0][1]);
    ag[1][0] = wmma_bf16(a1, bg0, ag[1][0]);
    ag[1][1] = wmma_bf16(a1, bg1, ag[1][1]);
  }

  float invs = amax[0] * amax[1 + e] * (1.f / (2688.f * 2688.f));  // 1/(gsf_x*gsf_w1)
  float lm = 0.f;
#pragma unroll
  for (int i = 0; i < 2; ++i)
#pragma unroll
    for (int j = 0; j < 2; ++j)
#pragma unroll
      for (int v = 0; v < 8; ++v) {
        int row = m0 + i * 16 + ((lane >> 4) << 3) + v;
        int col = n0 + j * 16 + (lane & 15);
        float val = av[i][j][v] * invs;
        float g   = ag[i][j][v] * invs;
        float h = 0.f;
        if (row - segstart < cnt_e) h = val * (g / (1.f + __expf(-g)));
        Hbuf[(size_t)row * kI + col] = h;
        lm = fmaxf(lm, fabsf(h));
      }
  for (int o = 16; o > 0; o >>= 1) lm = fmaxf(lm, __shfl_xor(lm, o, 32));
  if (lane == 0) atomicMax((unsigned int*)(amax + 17), __float_as_uint(lm));
}

__global__ __launch_bounds__(256) void k_qact(const float* __restrict__ hb,
                                              const int* __restrict__ ints,
                                              const float* __restrict__ amax, __bf16* a2) {
  int id = blockIdx.x * 256 + threadIdx.x;
  if (id >= kCap * (kI / 16)) return;
  int row = id / (kI / 16);
  if (row >= ints[16]) return;
  float gsf = 2688.f / bfr(amax[17]);
  __bf16 o[16] __attribute__((aligned(32)));
  nvfp4_block(hb + (size_t)id * 16, gsf, o);
  store16(a2 + (size_t)id * 16, o);
}

// gemm2: act @ w2^T, weighted scatter-add into out (exactly TOPK commutative adds/elem)
__global__ __launch_bounds__(256) void k_gemm2(const __bf16* __restrict__ A2,
                                               const __bf16* __restrict__ B2,
                                               const int* __restrict__ ints,
                                               const int* __restrict__ rowt,
                                               const float* __restrict__ roww,
                                               const float* __restrict__ amax,
                                               float* out) {
  __shared__ int s_off[9];
  int tid = threadIdx.x;
  if (tid < 9) s_off[tid] = (tid < 8) ? ints[8 + tid] : ints[16];
  __syncthreads();
  int mbase = blockIdx.y * 64;
  if (mbase >= s_off[8]) return;
  int e = 0;
  while (e < 7 && mbase >= s_off[e + 1]) ++e;
  int cnt_e = ints[e], segstart = s_off[e];

  int lane = tid & 31, w = tid >> 5;
  int wm = w & 1, wn = w >> 1;
  int m0 = mbase + wm * 32;
  int n0 = blockIdx.x * 128 + wn * 32;

  int koff = (lane >> 4) << 4;
  const __bf16* a0p = A2 + (size_t)(m0 + (lane & 15)) * kI + koff;
  const __bf16* a1p = a0p + (size_t)16 * kI;
  const __bf16* Bb  = B2 + (size_t)e * kH * kI;
  int nt = n0 >> 4;

  v8f acc[2][2];
#pragma unroll
  for (int i = 0; i < 2; ++i)
#pragma unroll
    for (int j = 0; j < 2; ++j) acc[i][j] = zero8();

  for (int kt = 0; kt < kI / 32; ++kt) {
    v16bf a0 = load_a(a0p + kt * 32);
    v16bf a1 = load_a(a1p + kt * 32);
    v16bf b0 = load_b(Bb, kt, nt,     kNT2, lane);
    v16bf b1 = load_b(Bb, kt, nt + 1, kNT2, lane);
    acc[0][0] = wmma_bf16(a0, b0, acc[0][0]);
    acc[0][1] = wmma_bf16(a0, b1, acc[0][1]);
    acc[1][0] = wmma_bf16(a1, b0, acc[1][0]);
    acc[1][1] = wmma_bf16(a1, b1, acc[1][1]);
  }

  float invs = bfr(amax[17]) * amax[9 + e] * (1.f / (2688.f * 2688.f)); // 1/(gsf_act*gsf_w2)
#pragma unroll
  for (int i = 0; i < 2; ++i)
#pragma unroll
    for (int j = 0; j < 2; ++j)
#pragma unroll
      for (int v = 0; v < 8; ++v) {
        int row = m0 + i * 16 + ((lane >> 4) << 3) + v;
        int col = n0 + j * 16 + (lane & 15);
        if (row - segstart < cnt_e) {
          int t = rowt[row] >> 1;
          float s = acc[i][j][v] * invs * roww[row];
          atomicAdd(out + (size_t)t * kH + col, s);
        }
      }
}

} // namespace

extern "C" void kernel_launch(void* const* d_in, const int* in_sizes, int n_in,
                              void* d_out, int out_size, void* d_ws, size_t ws_size,
                              hipStream_t stream) {
  if (ws_size < WS_NEED) return;
  const float* x   = (const float*)d_in[0];
  const float* rw  = (const float*)d_in[1];
  const float* w1  = (const float*)d_in[2];
  const float* w2  = (const float*)d_in[3];
  const int*   ri  = (const int*)d_in[4];
  float* out = (float*)d_out;

  char* ws = (char*)d_ws;
  float* amax = (float*)(ws + OFF_AMAX);
  int*   ints = (int*)(ws + OFF_INTS);
  int*   rowt = (int*)(ws + OFF_ROWT);
  float* roww = (float*)(ws + OFF_ROWW);
  __bf16* Xs  = (__bf16*)(ws + OFF_XS);
  __bf16* B1  = (__bf16*)(ws + OFF_B1);
  __bf16* B2  = (__bf16*)(ws + OFF_B2);
  float*  Hb  = (float*)(ws + OFF_HB);
  __bf16* A2  = (__bf16*)(ws + OFF_A2);

  k_init<<<(kT * kH + 255) / 256, 256, 0, stream>>>(out, amax);
  k_amax<<<128, 256, 0, stream>>>(x, kT * kH, 128, amax + 0);
  k_amax<<<kE * 128, 256, 0, stream>>>(w1, 2 * kI * kH, 128, amax + 1);
  k_amax<<<kE * 128, 256, 0, stream>>>(w2, kH * kI, 128, amax + 9);
  k_route<<<1, 64, 0, stream>>>(ri, rw, ints, rowt, roww);
  k_qx<<<(kT * kH / 16 + 255) / 256, 256, 0, stream>>>(x, amax, Xs);
  k_qw1<<<(kE * (kH / 16) * (2 * kI) + 255) / 256, 256, 0, stream>>>(w1, amax, B1);
  k_qw2<<<(kE * (kI / 16) * kH + 255) / 256, 256, 0, stream>>>(w2, amax, B2);
  k_gemm1<<<dim3(kI / 128, kCap / 64), 256, 0, stream>>>(Xs, B1, ints, rowt, amax, Hb);
  k_qact<<<(kCap * (kI / 16) + 255) / 256, 256, 0, stream>>>(Hb, ints, amax, A2);
  k_gemm2<<<dim3(kH / 128, kCap / 64), 256, 0, stream>>>(A2, B2, ints, rowt, roww, amax, out);
}
